// RNA3D_78245714199206
// MI455X (gfx1250) — matrix operand order repaired and checked by reading, not compiled, not run on hardware
//
#include <hip/hip_runtime.h>

// ---------------- Types for WMMA ----------------
typedef __attribute__((ext_vector_type(16))) __bf16        v16bf;
typedef __attribute__((ext_vector_type(8)))  float         v8f;
typedef __attribute__((ext_vector_type(4)))  unsigned int  u32x4;

#define F_OUT 512     // HEADS * HID
#define HEADS 8
#define HID   64
#define NEG_SLOPE 0.2f

union Frag { u32x4 q[2]; v16bf v; };   // 16 bf16 = two b128 chunks

// round-to-nearest-even f32 -> bf16 bits
__device__ __forceinline__ unsigned short f32_to_bf16(float f) {
    unsigned u = __float_as_uint(f);
    unsigned r = u + 0x7FFFu + ((u >> 16) & 1u);
    return (unsigned short)(r >> 16);
}

// order-preserving float<->u32 map for atomicMax-based segment max
__device__ __forceinline__ unsigned ford(float f) {
    unsigned u = __float_as_uint(f);
    return (u & 0x80000000u) ? ~u : (u ^ 0x80000000u);
}
__device__ __forceinline__ float fdec(unsigned s) {
    unsigned u = (s & 0x80000000u) ? (s ^ 0x80000000u) : ~s;
    return __uint_as_float(u);
}

// ---------------- weight pre-convert: Wb[512,Kp] = bf16(W[K,512])^T, zero-padded ----------------
__global__ void convert_w(const float* __restrict__ W, unsigned short* __restrict__ Wb,
                          int K, int Kp)
{
    int t = blockIdx.x * blockDim.x + threadIdx.x;
    if (t >= Kp * F_OUT) return;
    int k = t % Kp, n = t / Kp;
    Wb[t] = f32_to_bf16(k < K ? W[(size_t)k * F_OUT + n] : 0.f);
}

// ---------------- layer-0 activation convert: actb[Np,Kp] = bf16(x[N,K]), zero-padded ----------------
__global__ void convert_x(const float* __restrict__ x, unsigned short* __restrict__ actb,
                          int n, int np, int K, int Kp)
{
    int t = blockIdx.x * blockDim.x + threadIdx.x;
    if (t >= np * Kp) return;
    int k = t % Kp, node = t / Kp;
    actb[t] = f32_to_bf16((node < n && k < K) ? x[(size_t)node * K + k] : 0.f);
}

// ---------------- inter-layer epilogue fused into bf16 convert (rows >= n zero-padded) ----------------
// actb = bf16( relu( agg/(denom+eps) + bias ) )
__global__ void finalize_bf16(const float* __restrict__ agg, const float* __restrict__ denom,
                              const float* __restrict__ bias, unsigned short* __restrict__ actb,
                              int n, int np)
{
    int t = blockIdx.x * blockDim.x + threadIdx.x;
    if (t >= np * F_OUT) return;
    int c = t & (F_OUT - 1), node = t >> 9;
    float v = 0.f;
    if (node < n) {
        v = agg[t] / (denom[node * HEADS + (c >> 6)] + 1e-16f) + bias[c];
        v = v > 0.f ? v : 0.f;
    }
    actb[t] = f32_to_bf16(v);
}

// final-layer epilogue (no relu, keep f32 for the prediction head)
__global__ void finalize_f32(const float* __restrict__ agg, const float* __restrict__ denom,
                             const float* __restrict__ bias, float* __restrict__ out, int n)
{
    int t = blockIdx.x * blockDim.x + threadIdx.x;
    if (t >= n * F_OUT) return;
    int c = t & (F_OUT - 1), node = t >> 9;
    out[t] = agg[t] / (denom[node * HEADS + (c >> 6)] + 1e-16f) + bias[c];
}

// ---------------- WMMA GEMM: out[Mp,512] = actb[Mp,Kp](bf16) @ Wb[Kp,512](bf16) ----------------
// One wave computes a 32x64 strip: 2 (M) x 4 (N) 16x16 tiles, K stepped by 32.
// All operands row-padded (Mp%32==0, Kp%32==0): no guards, no exec manipulation anywhere.
__global__ __launch_bounds__(128)
void gemm_wmma(const unsigned short* __restrict__ A, const unsigned short* __restrict__ B,
               float* __restrict__ out, int Mp, int Kp)
{
    const int wid     = blockIdx.x * 4 + (threadIdx.x >> 5);
    const int lane    = threadIdx.x & 31;
    const int mBlocks = Mp >> 5;
    const int bm      = wid >> 3;          // 512/64 = 8 column groups
    const int bn      = wid & 7;
    if (bm >= mBlocks) return;             // wave-uniform exit: EXEC all-1 at WMMA
    const int row0 = bm << 5;
    const int col0 = bn << 6;

    v8f acc[2][4];
    #pragma unroll
    for (int mi = 0; mi < 2; ++mi)
        #pragma unroll
        for (int t = 0; t < 4; ++t) acc[mi][t] = v8f{};

    const int l15 = lane & 15;
    const int hi  = lane >> 4;

    // per-lane base pointers (ISA 16-bit A layout / B layout)
    const unsigned short* aBase0 = A + (size_t)(row0 + l15) * Kp + (hi << 3);
    const unsigned short* aBase1 = aBase0 + (size_t)16 * Kp;
    const unsigned short* bBase  = B + (size_t)(col0 + l15) * Kp + (hi << 3);

    for (int k0 = 0; k0 < Kp; k0 += 32) {
        Frag a0, a1;
        const u32x4* pa0 = (const u32x4*)(aBase0 + k0);
        const u32x4* pa1 = (const u32x4*)(aBase1 + k0);
        a0.q[0] = pa0[0];  a0.q[1] = pa0[2];   // K = kb..kb+7, kb+16..kb+23
        a1.q[0] = pa1[0];  a1.q[1] = pa1[2];

        Frag b[4];
        const u32x4* pb = (const u32x4*)(bBase + k0);
        #pragma unroll
        for (int t = 0; t < 4; ++t) {          // 4 tiles, 16 columns apart; K = kb..kb+7, kb+16..kb+23
            b[t].q[0] = pb[2 * t * Kp];
            b[t].q[1] = pb[2 * t * Kp + 2];
        }
        #pragma unroll
        for (int t = 0; t < 4; ++t) {
            acc[0][t] = __builtin_amdgcn_wmma_f32_16x16x32_bf16(
                false, a0.v, false, b[t].v, (short)0, acc[0][t], false, false);
            acc[1][t] = __builtin_amdgcn_wmma_f32_16x16x32_bf16(
                false, a1.v, false, b[t].v, (short)0, acc[1][t], false, false);
        }
    }
    // ---- store (C layout: VGPR r -> M = r + 8*(lane>=16), col = lane&15); out row-padded ----
    #pragma unroll
    for (int mi = 0; mi < 2; ++mi) {
        const int srow = row0 + (mi << 4) + (hi << 3);
        #pragma unroll
        for (int r = 0; r < 8; ++r) {
            float* o = out + (size_t)(srow + r) * F_OUT + col0 + l15;
            o[0]  = acc[mi][0][r];
            o[16] = acc[mi][1][r];
            o[32] = acc[mi][2][r];
            o[48] = acc[mi][3][r];
        }
    }
}

// ---------------- per-node alpha_src/alpha_dst + init of max/denom buffers ----------------
__global__ void node_alpha(const float* __restrict__ h,
                           const float* __restrict__ aS, const float* __restrict__ aD,
                           float* __restrict__ asrc, float* __restrict__ adst,
                           unsigned* __restrict__ mbuf, float* __restrict__ denom, int n)
{
    int t = blockIdx.x * blockDim.x + threadIdx.x;
    if (t >= n * HEADS) return;
    int hd = t & 7, node = t >> 3;
    const float4* hp = (const float4*)(h + (size_t)node * F_OUT + hd * HID);
    const float4* sp = (const float4*)(aS + hd * HID);
    const float4* dp = (const float4*)(aD + hd * HID);
    float s = 0.f, d = 0.f;
    #pragma unroll
    for (int j = 0; j < 16; ++j) {
        float4 hv = hp[j], sv = sp[j], dv = dp[j];
        s += hv.x*sv.x + hv.y*sv.y + hv.z*sv.z + hv.w*sv.w;
        d += hv.x*dv.x + hv.y*dv.y + hv.z*dv.z + hv.w*dv.w;
    }
    asrc[t] = s; adst[t] = d;
    mbuf[t] = 0u;        // ford-encoded: below every real value
    denom[t] = 0.f;
}

// ---------------- zero aggregation output ----------------
__global__ void zero_out(float* __restrict__ out, int n)
{
    int t = blockIdx.x * blockDim.x + threadIdx.x;
    if (t < n * F_OUT) out[t] = 0.f;
}

__device__ __forceinline__ void edge_sd(const int* ei, int E, int e, int& s, int& d)
{
    if (e < E) { s = ei[e]; d = ei[E + e]; } else { s = d = e - E; }  // self loops
}

// ---------------- segment max over dst (ordered-int atomicMax) ----------------
__global__ void edge_max(const int* __restrict__ ei, int E, int n,
                         const float* __restrict__ asrc, const float* __restrict__ adst,
                         unsigned* __restrict__ mbuf)
{
    int t = blockIdx.x * blockDim.x + threadIdx.x;
    if (t >= (E + n) * HEADS) return;
    int hd = t & 7, e = t >> 3, s, d;
    edge_sd(ei, E, e, s, d);
    float v = asrc[s * 8 + hd] + adst[d * 8 + hd];
    v = v > 0.f ? v : NEG_SLOPE * v;
    atomicMax(&mbuf[d * 8 + hd], ford(v));
}

// ---------------- per-edge weight w=exp(e-m), segment sum of w ----------------
__global__ void edge_sum(const int* __restrict__ ei, int E, int n,
                         const float* __restrict__ asrc, const float* __restrict__ adst,
                         const unsigned* __restrict__ mbuf, float* __restrict__ denom,
                         float* __restrict__ ew)
{
    int t = blockIdx.x * blockDim.x + threadIdx.x;
    if (t >= (E + n) * HEADS) return;
    int hd = t & 7, e = t >> 3, s, d;
    edge_sd(ei, E, e, s, d);
    int idx = d * 8 + hd;
    float v = asrc[s * 8 + hd] + adst[idx];
    v = v > 0.f ? v : NEG_SLOPE * v;
    float w = __expf(v - fdec(mbuf[idx]));
    ew[t] = w;
    atomicAdd(&denom[idx], w);
}

// ---------------- message aggregation (un-normalized): out[dst] += w * h[src] ----------------
// 32 threads per edge, 16 channels each (within one head).
__global__ void edge_agg(const int* __restrict__ ei, int E, int n,
                         const float* __restrict__ ew,
                         const float* __restrict__ h, float* __restrict__ out)
{
    int t = blockIdx.x * blockDim.x + threadIdx.x;
    if (t >= (E + n) * 32) return;
    int sub = t & 31, e = t >> 5, s, d;
    edge_sd(ei, E, e, s, d);
    int chan0 = sub << 4;            // 0..496
    float w = ew[e * 8 + (chan0 >> 6)];
    const float4* hp = (const float4*)(h + (size_t)s * F_OUT + chan0);
    float* op = out + (size_t)d * F_OUT + chan0;
    #pragma unroll
    for (int j = 0; j < 4; ++j) {
        float4 hv = hp[j];
        atomicAdd(op + 4*j + 0, hv.x * w);
        atomicAdd(op + 4*j + 1, hv.y * w);
        atomicAdd(op + 4*j + 2, hv.z * w);
        atomicAdd(op + 4*j + 3, hv.w * w);
    }
}

// ---------------- final prediction: out[N,3] = h @ predW + predb ----------------
__global__ void predict(const float* __restrict__ h, const float* __restrict__ Wp,
                        const float* __restrict__ bp, float* __restrict__ out, int n)
{
    int node = blockIdx.x * blockDim.x + threadIdx.x;
    if (node >= n) return;
    float c0 = bp[0], c1 = bp[1], c2 = bp[2];
    const float4* hp = (const float4*)(h + (size_t)node * F_OUT);
    for (int j = 0; j < F_OUT / 4; ++j) {
        float4 hv = hp[j];
        int k = j * 4;
        c0 += hv.x*Wp[k*3+0] + hv.y*Wp[(k+1)*3+0] + hv.z*Wp[(k+2)*3+0] + hv.w*Wp[(k+3)*3+0];
        c1 += hv.x*Wp[k*3+1] + hv.y*Wp[(k+1)*3+1] + hv.z*Wp[(k+2)*3+1] + hv.w*Wp[(k+3)*3+1];
        c2 += hv.x*Wp[k*3+2] + hv.y*Wp[(k+1)*3+2] + hv.z*Wp[(k+2)*3+2] + hv.w*Wp[(k+3)*3+2];
    }
    out[node*3+0] = c0; out[node*3+1] = c1; out[node*3+2] = c2;
}

extern "C" void kernel_launch(void* const* d_in, const int* in_sizes, int n_in,
                              void* d_out, int out_size, void* d_ws, size_t ws_size,
                              hipStream_t stream) {
    const float* x  = (const float*)d_in[0];
    const int*   ei = (const int*)d_in[1];
    const int Nn = in_sizes[2];            // batch is [N]
    const int K0 = in_sizes[0] / Nn;       // 69
    const int E  = in_sizes[1] / 2;
    const int Np = (Nn + 31) & ~31;        // row-padded M for guard-free WMMA

    const float* W[3]  = {(const float*)d_in[3],  (const float*)d_in[7],  (const float*)d_in[11]};
    const float* aS[3] = {(const float*)d_in[4],  (const float*)d_in[8],  (const float*)d_in[12]};
    const float* aD[3] = {(const float*)d_in[5],  (const float*)d_in[9],  (const float*)d_in[13]};
    const float* b[3]  = {(const float*)d_in[6],  (const float*)d_in[10], (const float*)d_in[14]};
    const float* predW = (const float*)d_in[15];
    const float* predb = (const float*)d_in[16];

    // workspace layout:
    // bufA[Np*512]f32 | bufB[N*512]f32 | asrc[N*8] | adst[N*8] | mbuf[N*8]u32 | denom[N*8]
    // | ew[(E+N)*8]f32 | actb[Np*512]bf16 | Wb[512*512]bf16
    float*          bufA  = (float*)d_ws;
    float*          bufB  = bufA + (size_t)Np * F_OUT;
    float*          asrc  = bufB + (size_t)Nn * F_OUT;
    float*          adst  = asrc + (size_t)Nn * HEADS;
    unsigned*       mbuf  = (unsigned*)(adst + (size_t)Nn * HEADS);
    float*          denom = (float*)(mbuf + (size_t)Nn * HEADS);
    float*          ew    = denom + (size_t)Nn * HEADS;
    unsigned short* actb  = (unsigned short*)(ew + (size_t)(E + Nn) * HEADS);
    unsigned short* Wb    = actb + (size_t)Np * F_OUT;

    const int tNode   = Nn * HEADS;
    const int tFeat   = Nn * F_OUT;
    const int tFeatP  = Np * F_OUT;
    const int tEdgeH  = (E + Nn) * HEADS;
    const int tEdge32 = (E + Nn) * 32;
    const int gemmBlocks = ((Np / 32) * 8 + 3) / 4;   // 4 waves/block, 8 col-groups

    for (int l = 0; l < 3; ++l) {
        const int K  = (l == 0) ? K0 : F_OUT;
        const int Kp = (K + 31) & ~31;              // 96 or 512

        // --- prepare bf16 operands (fused epilogue for l>0), pad rows zeroed ---
        if (l == 0) {
            int tc = Np * Kp;
            convert_x<<<(tc + 255) / 256, 256, 0, stream>>>(x, actb, Nn, Np, K, Kp);
        } else {
            finalize_bf16<<<(tFeatP + 255) / 256, 256, 0, stream>>>(bufB, denom, b[l-1], actb, Nn, Np);
        }
        int tw = Kp * F_OUT;
        convert_w<<<(tw + 255) / 256, 256, 0, stream>>>(W[l], Wb, K, Kp);

        // --- dense GEMM (WMMA bf16, f32 acc), guard-free ---
        gemm_wmma<<<gemmBlocks, 128, 0, stream>>>(actb, Wb, bufA, Np, Kp);

        // --- attention logits + buffer init ---
        node_alpha<<<(tNode + 255) / 256, 256, 0, stream>>>(bufA, aS[l], aD[l], asrc, adst, mbuf, denom, Nn);
        zero_out<<<(tFeat + 255) / 256, 256, 0, stream>>>(bufB, Nn);

        // --- segment softmax + aggregation ---
        edge_max<<<(tEdgeH + 255) / 256, 256, 0, stream>>>(ei, E, Nn, asrc, adst, mbuf);
        edge_sum<<<(tEdgeH + 255) / 256, 256, 0, stream>>>(ei, E, Nn, asrc, adst, mbuf, denom, ew);
        edge_agg<<<(tEdge32 + 255) / 256, 256, 0, stream>>>(ei, E, Nn, ew, bufA, bufB);
    }

    // final epilogue (divide + bias, no relu) then prediction head
    finalize_f32<<<(tFeat + 255) / 256, 256, 0, stream>>>(bufB, denom, b[2], bufA, Nn);
    predict<<<(Nn + 255) / 256, 256, 0, stream>>>(bufA, predW, predb, (float*)d_out, Nn);
}
